// LSTM_21543555956818
// MI455X (gfx1250) — matrix operand order repaired and checked by reading, not compiled, not run on hardware
//
#include <hip/hip_runtime.h>

typedef __attribute__((ext_vector_type(16))) _Float16 v16h;
typedef __attribute__((ext_vector_type(8)))  _Float16 v8h;
typedef __attribute__((ext_vector_type(8)))  float    v8f;

#define INPUT  128
#define HIDDEN 256
#define GATES  (4 * HIDDEN)   // 1024
#define SEQ    512
#define BATCH  512

// Workspace layout (bytes). Total ~2.3 MB, fully L2-resident.
#define OFF_WHH   (0u)                 // GATES*HIDDEN f16 = 512 KB
#define OFF_WIH   (512u * 1024u)       // GATES*INPUT  f16 = 256 KB
#define OFF_BIAS  (768u * 1024u)       // GATES f32         =   4 KB
#define OFF_H16_0 (784u * 1024u)       // BATCH*HIDDEN f16  = 256 KB
#define OFF_H16_1 (1040u * 1024u)      // BATCH*HIDDEN f16  = 256 KB
#define OFF_C     (1296u * 1024u)      // BATCH*HIDDEN f32  = 512 KB
#define OFF_HF32  (1808u * 1024u)      // BATCH*HIDDEN f32  = 512 KB

// ---------------------------------------------------------------------------
// Setup: f32 weights -> f16, bias = b_ih + b_hh, zero h0 (f16) and c0 (f32).
// Runs every kernel_launch call (no persistent state assumptions).
// ---------------------------------------------------------------------------
__global__ void lstm_setup(const float* __restrict__ Wih,
                           const float* __restrict__ Whh,
                           const float* __restrict__ bih,
                           const float* __restrict__ bhh,
                           _Float16* __restrict__ wih16,
                           _Float16* __restrict__ whh16,
                           float* __restrict__ bias,
                           _Float16* __restrict__ h16_0,
                           float* __restrict__ c) {
    int idx = blockIdx.x * blockDim.x + threadIdx.x;
    if (idx < GATES * HIDDEN) whh16[idx] = (_Float16)Whh[idx];
    if (idx < GATES * INPUT)  wih16[idx] = (_Float16)Wih[idx];
    if (idx < GATES)          bias[idx]  = bih[idx] + bhh[idx];
    if (idx < BATCH * HIDDEN) { h16_0[idx] = (_Float16)0.0f; c[idx] = 0.0f; }
}

// ---------------------------------------------------------------------------
// One LSTM timestep. Block = 128 threads = 4 waves; wave g computes the
// 16x16 tile of gate g (i,f,g,o) for batch tile blockIdx.x, hidden tile
// blockIdx.y, fusing x_t@Wih^T (K=128) and h@Whh^T (K=256) via
// v_wmma_f32_16x16x32_f16. Gates -> LDS -> activations -> c/h update.
// c updated in place (tile-private); h double-buffered across steps.
// ---------------------------------------------------------------------------
__global__ __launch_bounds__(128) void lstm_step(
    const float*    __restrict__ xt,     // (BATCH, INPUT) f32
    const _Float16* __restrict__ wih16,  // (GATES, INPUT)
    const _Float16* __restrict__ whh16,  // (GATES, HIDDEN)
    const float*    __restrict__ bias,   // (GATES)
    const _Float16* __restrict__ h_in,   // (BATCH, HIDDEN)
    _Float16*       __restrict__ h_out,  // (BATCH, HIDDEN)
    float*          __restrict__ c,      // (BATCH, HIDDEN)
    float*          __restrict__ hf32)   // (BATCH, HIDDEN)
{
    __shared__ float gbuf[4][16][17];    // +1 pad: avoid LDS bank conflicts

    const int bm   = blockIdx.x;             // batch tile 0..31
    const int hn   = blockIdx.y;             // hidden tile 0..15
    const int wave = threadIdx.x >> 5;       // gate index: 0=i 1=f 2=g 3=o
    const int lane = threadIdx.x & 31;
    const int lm   = lane & 15;              // M (A) / N (B,C) index
    const int lk   = lane >> 4;              // K-half select

    const int arow = bm * 16 + lm;                    // batch row for A frags
    const int j    = wave * HIDDEN + hn * 16 + lm;    // gate column for B/bias

    v8f acc = {};

    // ---- x_t contribution: K = 128 (4 chunks of 32), f32 -> f16 on the fly
    const float* xrow = xt + (size_t)arow * INPUT;
    #pragma unroll
    for (int kc = 0; kc < INPUT / 32; ++kc) {
        const int k0 = kc * 32;
        v16h a;
        const float* p0 = xrow + k0 + lk * 16;       // K = k0 + lk*16 + 0..7
        const float* p1 = xrow + k0 + 8 + lk * 16;   // K = k0+8 + lk*16 + 0..7
        #pragma unroll
        for (int i = 0; i < 8; ++i) {
            a[i]     = (_Float16)p0[i];
            a[8 + i] = (_Float16)p1[i];
        }
        v16h b = *(const v16h*)(wih16 + (size_t)j * INPUT + k0 + lk * 16);
        acc = __builtin_amdgcn_wmma_f32_16x16x32_f16(
                  false, a, false, b, (short)0, acc, false, false);
    }

    // ---- recurrent contribution: K = 256 (8 chunks of 32), f16 direct loads
    const _Float16* hrow = h_in + (size_t)arow * HIDDEN;
    #pragma unroll
    for (int kc = 0; kc < HIDDEN / 32; ++kc) {
        const int k0 = kc * 32;
        v8h a0 = *(const v8h*)(hrow + k0 + lk * 16);
        v8h a1 = *(const v8h*)(hrow + k0 + 8 + lk * 16);
        v16h a;
        #pragma unroll
        for (int i = 0; i < 8; ++i) { a[i] = a0[i]; a[8 + i] = a1[i]; }
        v16h b = *(const v16h*)(whh16 + (size_t)j * HIDDEN + k0 + lk * 16);
        acc = __builtin_amdgcn_wmma_f32_16x16x32_f16(
                  false, a, false, b, (short)0, acc, false, false);
    }

    // ---- C/D layout: VGPR r, lanes 0-15 -> M=r, lanes 16-31 -> M=8+r; N=lm
    const float bj = bias[j];
    #pragma unroll
    for (int r = 0; r < 8; ++r) {
        gbuf[wave][r + lk * 8][lm] = acc[r] + bj;
    }
    __syncthreads();

    // ---- activations + state update: 256 cells, 2 per thread
    #pragma unroll
    for (int e = 0; e < 2; ++e) {
        const int idx = threadIdx.x * 2 + e;
        const int m = idx >> 4, n = idx & 15;
        const float ig = 1.0f / (1.0f + __expf(-gbuf[0][m][n]));
        const float fg = 1.0f / (1.0f + __expf(-gbuf[1][m][n]));
        const float gg = tanhf(gbuf[2][m][n]);
        const float og = 1.0f / (1.0f + __expf(-gbuf[3][m][n]));
        const int gb = bm * 16 + m;
        const int gh = hn * 16 + n;
        const size_t off = (size_t)gb * HIDDEN + gh;
        const float cv = fg * c[off] + ig * gg;
        const float hv = og * tanhf(cv);
        c[off]     = cv;
        hf32[off]  = hv;
        h_out[off] = (_Float16)hv;
    }
}

// ---------------------------------------------------------------------------
// Final projection: out[b] = h_n[b,:] . W_out[0,:] + b_out. Tiny (512x256).
// ---------------------------------------------------------------------------
__global__ void lstm_proj(const float* __restrict__ hf32,
                          const float* __restrict__ Wout,
                          const float* __restrict__ bout,
                          float* __restrict__ out) {
    int b = blockIdx.x * blockDim.x + threadIdx.x;
    if (b >= BATCH) return;
    float s = bout[0];
    #pragma unroll 8
    for (int k = 0; k < HIDDEN; ++k) s += hf32[(size_t)b * HIDDEN + k] * Wout[k];
    out[b] = s;
}

extern "C" void kernel_launch(void* const* d_in, const int* in_sizes, int n_in,
                              void* d_out, int out_size, void* d_ws, size_t ws_size,
                              hipStream_t stream) {
    const float* x    = (const float*)d_in[0];  // (SEQ, BATCH, INPUT)
    const float* Wih  = (const float*)d_in[1];  // (4H, INPUT)
    const float* Whh  = (const float*)d_in[2];  // (4H, HIDDEN)
    const float* bih  = (const float*)d_in[3];  // (4H)
    const float* bhh  = (const float*)d_in[4];  // (4H)
    const float* Wout = (const float*)d_in[5];  // (1, HIDDEN)
    const float* bout = (const float*)d_in[6];  // (1)

    char* ws = (char*)d_ws;
    _Float16* whh16  = (_Float16*)(ws + OFF_WHH);
    _Float16* wih16  = (_Float16*)(ws + OFF_WIH);
    float*    bias   = (float*)   (ws + OFF_BIAS);
    _Float16* h16[2] = { (_Float16*)(ws + OFF_H16_0),
                         (_Float16*)(ws + OFF_H16_1) };
    float*    c      = (float*)   (ws + OFF_C);
    float*    hf32   = (float*)   (ws + OFF_HF32);

    lstm_setup<<<(GATES * HIDDEN + 255) / 256, 256, 0, stream>>>(
        Wih, Whh, bih, bhh, wih16, whh16, bias, h16[0], c);

    dim3 grid(BATCH / 16, HIDDEN / 16);   // 32 x 16 = 512 blocks / step
    for (int t = 0; t < SEQ; ++t) {
        lstm_step<<<grid, 128, 0, stream>>>(
            x + (size_t)t * BATCH * INPUT,
            wih16, whh16, bias,
            h16[t & 1], h16[(t + 1) & 1],
            c, hf32);
    }
    // after t=511 (reads buf1, writes buf0) the final hidden is in hf32
    lstm_proj<<<2, 256, 0, stream>>>(hf32, Wout, bout, (float*)d_out);
}
